// SwinTransformerBlock_71021579207119
// MI455X (gfx1250) — matrix operand order repaired and checked by reading, not compiled, not run on hardware
//
#include <hip/hip_runtime.h>
#include <hip/hip_bf16.h>
#include <math.h>

typedef __attribute__((ext_vector_type(16))) _Float16 v16h;
typedef __attribute__((ext_vector_type(8)))  float    v8f;

#define L_      2
#define D_      768
#define H_      12
#define DH_     64
#define FF_     3072
#define WS2_    49
#define MTOK    25088L   // 8*3136 tokens

union FragU { uint4 u2[2]; v16h v; };

// ISA A-layout (16x32 f16): lanes0-15 K=0..7 & 16..23, lanes16-31 K=8..15 & 24..31
__device__ __forceinline__ v16h load_afrag(const _Float16* rowbase, int hlf) {
    FragU f;
    f.u2[0] = *(const uint4*)(rowbase + hlf * 8);
    f.u2[1] = *(const uint4*)(rowbase + 16 + hlf * 8);
    return f.v;
}
// ISA B-layout (32x16 f16, N-major in LDS): lanes0-15 K=0..7 & 16..23, lanes16-31 K=8..15 & 24..31
__device__ __forceinline__ v16h load_bfrag(const _Float16* colbase, int hlf) {
    FragU f;
    f.u2[0] = *(const uint4*)(colbase + hlf * 8);
    f.u2[1] = *(const uint4*)(colbase + 16 + hlf * 8);
    return f.v;
}

// ---------------- GEMM: C = A(MxK,f16) * W(NxK,f16)^T  (+bias, epilogue) ----
// mode 0: out_f16 = acc + bias
// mode 1: hres_f32 += acc + bias      (residual accumulate)
// mode 2: out_f16 = gelu(acc + bias)  (exact GELU via erf)
#define BM 128
#define BN 128
#define BK 32
#define LDSP 40            // padded LDS row stride in halves (80B, conflict-free)
#define TILESZ (BM * LDSP) // halves per stage buffer

__global__ __launch_bounds__(256) void gemm_wmma_kernel(
    const _Float16* __restrict__ A, const _Float16* __restrict__ W,
    const float* __restrict__ bias, float* __restrict__ hres,
    _Float16* __restrict__ out, int M, int N, int K, int mode)
{
    // double-buffered stages
    __shared__ __align__(16) _Float16 As[2 * TILESZ];
    __shared__ __align__(16) _Float16 Bs[2 * TILESZ];

    const int tid   = threadIdx.x;
    const int wave  = tid >> 5;
    const int lane  = tid & 31;
    const int hlf   = lane >> 4;     // half-wave select
    const int r16   = lane & 15;
    const int wm    = wave & 3;      // 4 wave-rows -> 4*32 = 128 M
    const int wn    = wave >> 2;     // 2 wave-cols -> 2*64 = 128 N
    const long blockM = (long)blockIdx.y * BM;
    const long blockN = (long)blockIdx.x * BN;

    v8f acc[2][4];
#pragma unroll
    for (int mi = 0; mi < 2; ++mi)
#pragma unroll
        for (int ni = 0; ni < 4; ++ni)
#pragma unroll
            for (int j = 0; j < 8; ++j) acc[mi][ni][j] = 0.0f;

    const int srow = tid >> 1, seg = tid & 1;   // 128 rows x 2 segs of 16 halves
    const _Float16* gA = A + (blockM + srow) * (long)K + seg * 16;
    const _Float16* gB = W + (blockN + srow) * (long)K + seg * 16;
    const int sOff = srow * LDSP + seg * 16;    // halves, 16B-aligned
    const int nk = K / BK;

    // ---- prologue: stage tile 0 into buffer 0 ------------------------------
    {
        uint4 a0 = ((const uint4*)gA)[0];
        uint4 a1 = ((const uint4*)gA)[1];
        uint4 b0 = ((const uint4*)gB)[0];
        uint4 b1 = ((const uint4*)gB)[1];
        uint4* sA = (uint4*)(As + sOff);
        uint4* sB = (uint4*)(Bs + sOff);
        sA[0] = a0; sA[1] = a1;
        sB[0] = b0; sB[1] = b1;
        if (nk > 1) {
            __builtin_prefetch((const void*)(gA + BK), 0, 1);
            __builtin_prefetch((const void*)(gB + BK), 0, 1);
        }
    }
    __syncthreads();

    // ---- pipelined main loop: one barrier per K step -----------------------
    for (int kt = 0; kt < nk; ++kt) {
        const int cur = kt & 1;
        const int nxt = cur ^ 1;
        const bool has = (kt + 1 < nk);

        // issue next tile's global loads NOW (consumed after the WMMA burst)
        uint4 a0{}, a1{}, b0{}, b1{};
        if (has) {
            const _Float16* pA = gA + (kt + 1) * BK;
            const _Float16* pB = gB + (kt + 1) * BK;
            a0 = ((const uint4*)pA)[0];
            a1 = ((const uint4*)pA)[1];
            b0 = ((const uint4*)pB)[0];
            b1 = ((const uint4*)pB)[1];
            if (kt + 2 < nk) {
                __builtin_prefetch((const void*)(gA + (kt + 2) * BK), 0, 1);
                __builtin_prefetch((const void*)(gB + (kt + 2) * BK), 0, 1);
            }
        }

        // compute from current buffer
        const _Float16* Ac = As + cur * TILESZ;
        const _Float16* Bc = Bs + cur * TILESZ;
        v16h af[2], bf[4];
#pragma unroll
        for (int mi = 0; mi < 2; ++mi)
            af[mi] = load_afrag(Ac + (wm * 32 + mi * 16 + r16) * LDSP, hlf);
#pragma unroll
        for (int ni = 0; ni < 4; ++ni)
            bf[ni] = load_bfrag(Bc + (wn * 64 + ni * 16 + r16) * LDSP, hlf);
#pragma unroll
        for (int mi = 0; mi < 2; ++mi)
#pragma unroll
            for (int ni = 0; ni < 4; ++ni)
                acc[mi][ni] = __builtin_amdgcn_wmma_f32_16x16x32_f16(
                    false, af[mi], false, bf[ni],
                    (short)0, acc[mi][ni], false, false);

        // now commit next tile to the other buffer (waits happen here)
        if (has) {
            uint4* sA = (uint4*)(As + nxt * TILESZ + sOff);
            uint4* sB = (uint4*)(Bs + nxt * TILESZ + sOff);
            sA[0] = a0; sA[1] = a1;
            sB[0] = b0; sB[1] = b1;
        }
        __syncthreads();
    }

    // epilogue: C VGPR j -> row = j + 8*hlf, col = r16 within a 16x16 tile
#pragma unroll
    for (int mi = 0; mi < 2; ++mi) {
#pragma unroll
        for (int ni = 0; ni < 4; ++ni) {
            const long tm = blockM + wm * 32 + mi * 16;
            const long tn = blockN + wn * 64 + ni * 16;
#pragma unroll
            for (int j = 0; j < 8; ++j) {
                const long row = tm + j + 8 * hlf;
                const long col = tn + r16;
                float v = acc[mi][ni][j] + bias[col];
                const long idx = row * (long)N + col;
                if (mode == 1) {
                    hres[idx] += v;
                } else if (mode == 2) {
                    out[idx] = (_Float16)(0.5f * v * (1.0f + erff(v * 0.70710678118654752f)));
                } else {
                    out[idx] = (_Float16)v;
                }
            }
        }
    }
}

// ---------------- LayerNorm row kernel: y_f16 = (h-mu)/sigma * w + b ---------
__device__ __forceinline__ float wave_sum(float v) {
#pragma unroll
    for (int m = 16; m > 0; m >>= 1) v += __shfl_xor(v, m, 32);
    return v;
}

__global__ __launch_bounds__(256) void ln_f16_kernel(
    const float* __restrict__ h, const float* __restrict__ w,
    const float* __restrict__ b, _Float16* __restrict__ y)
{
    __shared__ float part[8];
    const int tid = threadIdx.x;
    const int wave = tid >> 5, lane = tid & 31;
    const long row = blockIdx.x;
    const float* hp = h + row * (long)D_;

    float v[3];
#pragma unroll
    for (int i = 0; i < 3; ++i) v[i] = hp[tid + i * 256];

    float s = wave_sum(v[0] + v[1] + v[2]);
    if (lane == 0) part[wave] = s;
    __syncthreads();
    float tot = 0.0f;
#pragma unroll
    for (int i = 0; i < 8; ++i) tot += part[i];
    const float mean = tot * (1.0f / (float)D_);
    __syncthreads();

    float sq = 0.0f;
#pragma unroll
    for (int i = 0; i < 3; ++i) { float d = v[i] - mean; sq += d * d; }
    sq = wave_sum(sq);
    if (lane == 0) part[wave] = sq;
    __syncthreads();
    float vtot = 0.0f;
#pragma unroll
    for (int i = 0; i < 8; ++i) vtot += part[i];
    const float inv = rsqrtf(vtot * (1.0f / (float)D_) + 1e-5f);

    _Float16* yp = y + row * (long)D_;
#pragma unroll
    for (int i = 0; i < 3; ++i) {
        const int c = tid + i * 256;
        yp[c] = (_Float16)((v[i] - mean) * inv * w[c] + b[c]);
    }
}

// ---------------- Windowed attention, WMMA: one block per (window, head) ----
// 49x49 attention padded to 64x64. 128 threads = 4 waves; wave w owns the
// 16-row M strip [16w, 16w+16). S = Q*K^T on WMMA, softmax via LDS (f32),
// O = P*V on WMMA with V staged transposed.
#define ATS 72   // LDS row stride in halves (144B: 16B-aligned, conflict-free)

__global__ __launch_bounds__(128) void attn_wmma_kernel(
    const _Float16* __restrict__ qkv, _Float16* __restrict__ ao)
{
    __shared__ __align__(16) _Float16 qs[64 * ATS];
    __shared__ __align__(16) _Float16 kx[64 * ATS];  // K row-major == B N-major
    __shared__ __align__(16) _Float16 ps[64 * ATS];  // probabilities (A layout)
    __shared__ __align__(16) _Float16 vt[64 * ATS];  // V transposed  (B layout)
    __shared__ float sc[64 * 68];                    // f32 scores

    const int hh   = blockIdx.x % H_;
    const long win = blockIdx.x / H_;
    const long t0  = win * WS2_;
    const int tid  = threadIdx.x;
    const int wave = tid >> 5, lane = tid & 31;
    const int hlf  = lane >> 4, r16 = lane & 15;

    const _Float16* base = qkv + t0 * (long)(3 * D_) + hh * DH_;

    // stage Q and K rows (b128), zero-padding rows 49..63
    for (int i = tid; i < 64 * 8; i += 128) {
        const int row = i >> 3, s8 = i & 7;
        uint4 zq = {0u, 0u, 0u, 0u}, zk = {0u, 0u, 0u, 0u};
        if (row < WS2_) {
            zq = *(const uint4*)(base + row * (long)(3 * D_) + s8 * 8);
            zk = *(const uint4*)(base + row * (long)(3 * D_) + D_ + s8 * 8);
        }
        *(uint4*)(qs + row * ATS + s8 * 8) = zq;
        *(uint4*)(kx + row * ATS + s8 * 8) = zk;
    }
    // stage V transposed: vt[d][j] = V[j][d], zero j>=49
    for (int i = tid; i < 64 * 64; i += 128) {
        const int j = i >> 6, d = i & 63;
        _Float16 v = (_Float16)0.0f;
        if (j < WS2_) v = base[j * (long)(3 * D_) + 2 * D_ + d];
        vt[d * ATS + j] = v;
    }
    __syncthreads();

    // ---- S = Q * K^T (scaled), wave computes 16x64 strip -------------------
    {
        v8f acc[4];
#pragma unroll
        for (int n = 0; n < 4; ++n)
#pragma unroll
            for (int j = 0; j < 8; ++j) acc[n][j] = 0.0f;

        const _Float16* arow = qs + (wave * 16 + r16) * ATS;
#pragma unroll
        for (int kk = 0; kk < DH_; kk += 32) {
            v16h a = load_afrag(arow + kk, hlf);
#pragma unroll
            for (int n = 0; n < 4; ++n) {
                v16h b = load_bfrag(kx + (n * 16 + r16) * ATS + kk, hlf);
                acc[n] = __builtin_amdgcn_wmma_f32_16x16x32_f16(
                    false, a, false, b, (short)0, acc[n], false, false);
            }
        }
#pragma unroll
        for (int n = 0; n < 4; ++n)
#pragma unroll
            for (int j = 0; j < 8; ++j)
                sc[(wave * 16 + j + 8 * hlf) * 68 + n * 16 + r16] = acc[n][j] * 0.125f;
    }
    __syncthreads();

    // ---- softmax over the 49 valid keys, emit f16 P (zero-padded) ----------
    if (tid < 64) {
        const float* srow = sc + tid * 68;
        float m = -1e30f;
        for (int j = 0; j < WS2_; ++j) m = fmaxf(m, srow[j]);
        float sum = 0.0f;
        for (int j = 0; j < WS2_; ++j) sum += __expf(srow[j] - m);
        const float rs = 1.0f / sum;
        _Float16* prow = ps + tid * ATS;
        for (int j = 0; j < WS2_; ++j) prow[j] = (_Float16)(__expf(srow[j] - m) * rs);
        for (int j = WS2_; j < 64; ++j) prow[j] = (_Float16)0.0f;
    }
    __syncthreads();

    // ---- O = P * V, wave computes 16x64 strip ------------------------------
    {
        v8f acc[4];
#pragma unroll
        for (int n = 0; n < 4; ++n)
#pragma unroll
            for (int j = 0; j < 8; ++j) acc[n][j] = 0.0f;

        const _Float16* arow = ps + (wave * 16 + r16) * ATS;
#pragma unroll
        for (int kk = 0; kk < 64; kk += 32) {
            v16h a = load_afrag(arow + kk, hlf);
#pragma unroll
            for (int n = 0; n < 4; ++n) {
                v16h b = load_bfrag(vt + (n * 16 + r16) * ATS + kk, hlf);
                acc[n] = __builtin_amdgcn_wmma_f32_16x16x32_f16(
                    false, a, false, b, (short)0, acc[n], false, false);
            }
        }
#pragma unroll
        for (int n = 0; n < 4; ++n)
#pragma unroll
            for (int j = 0; j < 8; ++j) {
                const int q = wave * 16 + j + 8 * hlf;
                if (q < WS2_)
                    ao[(t0 + q) * (long)D_ + hh * DH_ + n * 16 + r16] =
                        (_Float16)acc[n][j];
            }
    }
}

// ---------------- f32 -> f16 weight conversion ------------------------------
__global__ __launch_bounds__(256) void cvt_f16_kernel(
    const float* __restrict__ src, _Float16* __restrict__ dst, long n)
{
    const long i = (long)blockIdx.x * 256 + threadIdx.x;
    if (i < n) dst[i] = (_Float16)src[i];
}

// ---------------------------------------------------------------------------
extern "C" void kernel_launch(void* const* d_in, const int* in_sizes, int n_in,
                              void* d_out, int out_size, void* d_ws, size_t ws_size,
                              hipStream_t stream) {
    const float* x    = (const float*)d_in[0];
    const float* Wqkv = (const float*)d_in[1];
    const float* bqkv = (const float*)d_in[2];
    const float* Wo   = (const float*)d_in[3];
    const float* bo   = (const float*)d_in[4];
    const float* ln1w = (const float*)d_in[5];
    const float* ln1b = (const float*)d_in[6];
    const float* W1   = (const float*)d_in[7];
    const float* b1   = (const float*)d_in[8];
    const float* W2   = (const float*)d_in[9];
    const float* b2   = (const float*)d_in[10];
    const float* ln2w = (const float*)d_in[11];
    const float* ln2b = (const float*)d_in[12];

    float* h = (float*)d_out;   // running residual stream (fp32), 25088 x 768

    // workspace layout (f16)
    const long nWqkv = (long)L_ * 3 * D_ * D_;
    const long nWo   = (long)L_ * D_ * D_;
    const long nW1   = (long)L_ * FF_ * D_;
    const long nW2   = (long)L_ * D_ * FF_;
    _Float16* fWqkv = (_Float16*)d_ws;
    _Float16* fWo   = fWqkv + nWqkv;
    _Float16* fW1   = fWo   + nWo;
    _Float16* fW2   = fW1   + nW1;
    _Float16* bufY   = fW2   + nW2;            // 25088 x 768  (LN output)
    _Float16* bufQKV = bufY   + MTOK * D_;     // 25088 x 2304
    _Float16* bufAO  = bufQKV + MTOK * 3 * D_; // 25088 x 768  (attention out)
    _Float16* bufG   = bufAO  + MTOK * D_;     // 25088 x 3072 (gelu out)

    cvt_f16_kernel<<<(int)((nWqkv + 255) / 256), 256, 0, stream>>>(Wqkv, fWqkv, nWqkv);
    cvt_f16_kernel<<<(int)((nWo   + 255) / 256), 256, 0, stream>>>(Wo,   fWo,   nWo);
    cvt_f16_kernel<<<(int)((nW1   + 255) / 256), 256, 0, stream>>>(W1,   fW1,   nW1);
    cvt_f16_kernel<<<(int)((nW2   + 255) / 256), 256, 0, stream>>>(W2,   fW2,   nW2);

    // h = x
    hipMemcpyAsync(h, x, MTOK * D_ * sizeof(float), hipMemcpyDeviceToDevice, stream);

    const int gM = (int)(MTOK / BM);   // 196
    for (int l = 0; l < L_; ++l) {
        // --- attention sub-block ---
        ln_f16_kernel<<<(int)MTOK, 256, 0, stream>>>(h, ln1w + l * D_, ln1b + l * D_, bufY);
        gemm_wmma_kernel<<<dim3(3 * D_ / BN, gM), 256, 0, stream>>>(
            bufY, fWqkv + (long)l * 3 * D_ * D_, bqkv + l * 3 * D_,
            nullptr, bufQKV, (int)MTOK, 3 * D_, D_, 0);
        attn_wmma_kernel<<<(int)(MTOK / WS2_) * H_, 128, 0, stream>>>(bufQKV, bufAO);
        gemm_wmma_kernel<<<dim3(D_ / BN, gM), 256, 0, stream>>>(
            bufAO, fWo + (long)l * D_ * D_, bo + l * D_,
            h, nullptr, (int)MTOK, D_, D_, 1);
        // --- FFN sub-block ---
        ln_f16_kernel<<<(int)MTOK, 256, 0, stream>>>(h, ln2w + l * D_, ln2b + l * D_, bufY);
        gemm_wmma_kernel<<<dim3(FF_ / BN, gM), 256, 0, stream>>>(
            bufY, fW1 + (long)l * FF_ * D_, b1 + l * FF_,
            nullptr, bufG, (int)MTOK, FF_, D_, 2);
        gemm_wmma_kernel<<<dim3(D_ / BN, gM), 256, 0, stream>>>(
            bufG, fW2 + (long)l * D_ * FF_, b2 + l * D_,
            h, nullptr, (int)MTOK, D_, FF_, 1);
    }
}
